// Attention_71141838291005
// MI455X (gfx1250) — matrix operand order repaired and checked by reading, not compiled, not run on hardware
//
#include <hip/hip_runtime.h>
#include <hip/hip_bf16.h>
#include <stdint.h>

#define DEV __device__ __forceinline__

constexpr int CB   = 4;          // batch
constexpr int CS   = 2048;       // sequence
constexpr int CD   = 2048;       // model dim
constexpr int CNH  = 16;         // query heads
constexpr int CNKV = 4;          // kv heads
constexpr int CHD  = 128;        // head dim
constexpr int CM   = CB * CS;    // 8192 token rows
constexpr int CQD  = CNH * CHD;  // 2048
constexpr int CKVD = CNKV * CHD; // 512

typedef __attribute__((ext_vector_type(16))) __bf16   bf16x16;
typedef __attribute__((ext_vector_type(8)))  float    f32x8;
typedef __attribute__((ext_vector_type(4)))  uint32_t u32x4;
typedef __attribute__((ext_vector_type(8)))  int      i32x8;
typedef __attribute__((ext_vector_type(4)))  int      i32x4;

union Frag { bf16x16 bf; uint32_t u[8]; uint16_t h[16]; };

DEV uint16_t f2bf(float f) {
    union { float f; uint32_t u; } c; c.f = f;
    uint32_t u = c.u + 0x7FFFu + ((c.u >> 16) & 1u);   // round-to-nearest-even
    return (uint16_t)(u >> 16);
}
DEV float bf2f(uint16_t h) {
    union { uint32_t u; float f; } c; c.u = ((uint32_t)h) << 16;
    return c.f;
}

// ---------------------------------------------------------------- TDM helper
// 2D tile load via Tensor Data Mover: gptr -> LDS, with per-row LDS padding.
// dims/strides in elements (data_size = 2 bytes). One wave issues; EXEC ignored.
DEV void tdm_load_2d_bf16(uint32_t lds_addr, const uint16_t* gptr,
                          uint32_t tensor_d0, uint32_t tensor_d1, uint32_t stride0,
                          uint32_t tile_d0, uint32_t tile_d1,
                          uint32_t pad_interval_code, uint32_t pad_amount_code)
{
    uint64_t ga = (uint64_t)(uintptr_t)gptr;
    u32x4 g0;
    g0[0] = 1u;                                           // count=1 (valid user D#)
    g0[1] = lds_addr;                                     // LDS byte address
    g0[2] = (uint32_t)ga;                                 // global_addr[31:0]
    g0[3] = (uint32_t)((ga >> 32) & 0x01FFFFFFu) | (2u << 30);  // addr[56:32] | type=2
    i32x8 g1;
    g1[0] = (int)((1u << 16) |                            // data_size = 2 bytes
                  (1u << 20) |                            // pad_enable
                  (pad_interval_code << 22) |
                  (pad_amount_code   << 25));
    g1[1] = (int)((tensor_d0 & 0xFFFFu) << 16);           // tensor_dim0[15:0]
    g1[2] = (int)((tensor_d0 >> 16) | ((tensor_d1 & 0xFFFFu) << 16));
    g1[3] = (int)((tensor_d1 >> 16) | (tile_d0 << 16));   // | tile_dim0
    g1[4] = (int)tile_d1;                                 // tile_dim1, tile_dim2=0
    g1[5] = (int)stride0;                                 // tensor_dim0_stride lo
    g1[6] = 0;
    g1[7] = 0;
    i32x4 z4 = {0, 0, 0, 0};
#if defined(__clang_major__) && (__clang_major__ >= 23)
    i32x8 z8 = {0, 0, 0, 0, 0, 0, 0, 0};
    __builtin_amdgcn_tensor_load_to_lds(g0, g1, z4, z4, z8, 0);
#else
    __builtin_amdgcn_tensor_load_to_lds(g0, g1, z4, z4, 0);
#endif
}

// ---------------------------------------------------------------- fp32 -> bf16
__global__ void cvt_kernel(const float* __restrict__ in, uint16_t* __restrict__ out, int n)
{
    int i = (blockIdx.x * blockDim.x + threadIdx.x) * 4;
    if (i + 3 < n) {
        float4 v = *reinterpret_cast<const float4*>(in + i);
        uint2 o;
        o.x = (uint32_t)f2bf(v.x) | ((uint32_t)f2bf(v.y) << 16);
        o.y = (uint32_t)f2bf(v.z) | ((uint32_t)f2bf(v.w) << 16);
        *reinterpret_cast<uint2*>(out + i) = o;
    }
}

// ---------------------------------------------------------------- RoPE in-place
__global__ void rope_kernel(uint16_t* __restrict__ t, const float* __restrict__ cs,
                            const float* __restrict__ sn, int nheads)
{
    int i = blockIdx.x * blockDim.x + threadIdx.x;
    int total = CB * CS * nheads * (CHD / 2);
    if (i >= total) return;
    int p   = i & 63;
    int row = i >> 6;                       // (b*S + s)*nheads + h
    int s   = (row / nheads) % CS;
    size_t base = (size_t)row * CHD + 2 * p;
    float c  = cs[s * 64 + p];
    float sv = sn[s * 64 + p];
    float xr = bf2f(t[base]), xi = bf2f(t[base + 1]);
    t[base]     = f2bf(xr * c - xi * sv);
    t[base + 1] = f2bf(xr * sv + xi * c);
}

// ---------------------------------------------------------------- bf16 GEMM
// C[M,N] = A[M,K] * B[K,N]; A,B bf16 row-major; C f32 or bf16.
// 128x128x32 tile, 256 threads = 8 waves (4x2), each wave a 32x64 tile.
// Double-buffered LDS; A staged by TDM (wave 0), B transposed through registers.
template <bool OUT_F32>
__global__ __launch_bounds__(256)
void gemm_kernel(const uint16_t* __restrict__ A, const uint16_t* __restrict__ B,
                 void* __restrict__ Cp, int M, int N, int K)
{
    constexpr int BM = 128, BN = 128, BK = 32;
    constexpr int SAS  = BK + 8;    // A rows padded: 64B data + 16B pad (TDM pads)
    constexpr int SBTS = BK + 8;    // B^T rows padded likewise
    __shared__ uint16_t sA[2][BM * SAS];
    __shared__ uint16_t sBT[2][BN * SBTS];

    const int tid  = threadIdx.x;
    const int lane = tid & 31, wid = tid >> 5;
    const int lg = lane >> 4, ll = lane & 15;
    const int wm = wid >> 1, wn = wid & 1;
    const int m0 = blockIdx.y * BM, n0 = blockIdx.x * BN;

    f32x8 acc[2][4];
#pragma unroll
    for (int mi = 0; mi < 2; ++mi)
#pragma unroll
        for (int nt = 0; nt < 4; ++nt)
#pragma unroll
            for (int r = 0; r < 8; ++r) acc[mi][nt][r] = 0.f;

    const int br = tid >> 3, bc = (tid & 7) * 16;   // B stage: 32 rows x 128 cols

    // ---- prologue: stage k0=0 into buffer 0
    if (wid == 0)
        tdm_load_2d_bf16((uint32_t)(uintptr_t)&sA[0][0], A + (size_t)m0 * K,
                         (uint32_t)K, (uint32_t)BM, (uint32_t)K, BK, BM,
                         /*interval: 8 qwords = 64B rows*/ 3, /*pad: 4 dwords*/ 3);
    uint32_t w[8];
    {
        const uint16_t* src = B + (size_t)br * N + (n0 + bc);
        uint4 p0 = *reinterpret_cast<const uint4*>(src);
        uint4 p1 = *reinterpret_cast<const uint4*>(src + 8);
        w[0]=p0.x; w[1]=p0.y; w[2]=p0.z; w[3]=p0.w;
        w[4]=p1.x; w[5]=p1.y; w[6]=p1.z; w[7]=p1.w;
#pragma unroll
        for (int e = 0; e < 16; ++e)
            sBT[0][(bc + e) * SBTS + br] = (uint16_t)(w[e >> 1] >> ((e & 1) * 16));
    }
    if (wid == 0) __builtin_amdgcn_s_wait_tensorcnt(0);
    __syncthreads();

    int buf = 0;
    for (int k0 = 0; k0 < K; k0 += BK) {
        const int nxt = k0 + BK;
        const bool more = nxt < K;
        if (more) {   // prefetch next tile into buf^1
            if (wid == 0)
                tdm_load_2d_bf16((uint32_t)(uintptr_t)&sA[buf ^ 1][0],
                                 A + (size_t)m0 * K + nxt,
                                 (uint32_t)K, (uint32_t)BM, (uint32_t)K, BK, BM, 3, 3);
            const uint16_t* src = B + (size_t)(nxt + br) * N + (n0 + bc);
            uint4 p0 = *reinterpret_cast<const uint4*>(src);
            uint4 p1 = *reinterpret_cast<const uint4*>(src + 8);
            w[0]=p0.x; w[1]=p0.y; w[2]=p0.z; w[3]=p0.w;
            w[4]=p1.x; w[5]=p1.y; w[6]=p1.z; w[7]=p1.w;
        }

        // ---- compute from buf
        Frag a[2], b[4];
#pragma unroll
        for (int mi = 0; mi < 2; ++mi) {   // A: K = 16*lg + e contiguous (matches B)
            const uint16_t* base = &sA[buf][(wm * 32 + mi * 16 + ll) * SAS + lg * 16];
#pragma unroll
            for (int j = 0; j < 8; ++j) {
                int off = 2 * j;
                a[mi].u[j] = *reinterpret_cast<const uint32_t*>(base + off);
            }
        }
#pragma unroll
        for (int nt = 0; nt < 4; ++nt) {   // B^T: N = lane row, K = 16*lg + e contiguous
            const uint16_t* base = &sBT[buf][(wn * 64 + nt * 16 + ll) * SBTS + lg * 16];
#pragma unroll
            for (int j = 0; j < 8; ++j)
                b[nt].u[j] = *reinterpret_cast<const uint32_t*>(base + 2 * j);
        }
#pragma unroll
        for (int mi = 0; mi < 2; ++mi)
#pragma unroll
            for (int nt = 0; nt < 4; ++nt)
                acc[mi][nt] = __builtin_amdgcn_wmma_f32_16x16x32_bf16(
                    false, a[mi].bf, false, b[nt].bf, (short)0, acc[mi][nt], false, false);

        if (more) {   // finish staging B^T into buf^1 (A is in flight via TDM)
#pragma unroll
            for (int e = 0; e < 16; ++e)
                sBT[buf ^ 1][(bc + e) * SBTS + br] = (uint16_t)(w[e >> 1] >> ((e & 1) * 16));
        }
        if (wid == 0) __builtin_amdgcn_s_wait_tensorcnt(0);
        __syncthreads();
        buf ^= 1;
    }

    // C layout: element r -> M = r + 8*(lane>=16), N = lane&15
#pragma unroll
    for (int mi = 0; mi < 2; ++mi)
#pragma unroll
        for (int nt = 0; nt < 4; ++nt)
#pragma unroll
            for (int r = 0; r < 8; ++r) {
                int row = m0 + wm * 32 + mi * 16 + lg * 8 + r;
                int col = n0 + wn * 64 + nt * 16 + ll;
                if (OUT_F32)
                    reinterpret_cast<float*>(Cp)[(size_t)row * N + col] = acc[mi][nt][r];
                else
                    reinterpret_cast<uint16_t*>(Cp)[(size_t)row * N + col] = f2bf(acc[mi][nt][r]);
            }
}

// ---------------------------------------------------------------- flash attention
// Q bf16 [CB*CS, CNH*CHD], K/V bf16 [CB*CS, CNKV*CHD], O bf16 [CB*CS, CNH*CHD].
// 128 threads = 4 waves; each wave owns 16 query rows; 64-key blocks shared via LDS.
// K tile staged with async-to-LDS copies; V transposed through registers.
__global__ __launch_bounds__(128)
void flash_kernel(const uint16_t* __restrict__ Q, const uint16_t* __restrict__ K,
                  const uint16_t* __restrict__ V, uint16_t* __restrict__ O)
{
    constexpr int BC  = 64;
    constexpr int SKS = CHD + 8;   // sK  [key][d]   (QK^T B-frags read contiguous d)
    constexpr int SVS = BC + 8;    // sVT [d][key]   (PV  B-frags read contiguous key)
    constexpr int SPS = BC + 8;    // per-wave P bounce (C-layout -> A-layout)
    __shared__ uint16_t sK[BC * SKS];
    __shared__ uint16_t sVT[CHD * SVS];
    __shared__ uint16_t sP[4 * 16 * SPS];

    const int tid  = threadIdx.x;
    const int lane = tid & 31, wid = tid >> 5;
    const int lg = lane >> 4, ll = lane & 15;

    const int qblk = blockIdx.x;
    const int h    = blockIdx.y;
    const int b    = blockIdx.z;
    const int kvh  = h / (CNH / CNKV);
    const int q0   = qblk * 64 + wid * 16;     // this wave's 16 query rows

    // Q frags live in registers for the whole K loop (4 x 16x32 over d=128)
    Frag qf[4];
    {
        const size_t rowbase = ((size_t)(b * CS + q0 + ll)) * CQD + h * CHD;
#pragma unroll
        for (int kd = 0; kd < 4; ++kd) {
            const uint16_t* base = Q + rowbase + kd * 32 + lg * 16;
#pragma unroll
            for (int j = 0; j < 8; ++j) {
                int off = 2 * j;
                qf[kd].u[j] = *reinterpret_cast<const uint32_t*>(base + off);
            }
        }
    }

    float mrow[8], lrow[8];
    f32x8 o[8];
#pragma unroll
    for (int r = 0; r < 8; ++r) { mrow[r] = -1e30f; lrow[r] = 0.f; }
#pragma unroll
    for (int dt = 0; dt < 8; ++dt)
#pragma unroll
        for (int r = 0; r < 8; ++r) o[dt][r] = 0.f;

    const float scale = 0.08838834764831845f;  // 1/sqrt(128)
    const int kend = qblk * 64 + 64;           // causal: no keys past this block
    const unsigned long long kbase64 = (unsigned long long)(uintptr_t)K;

    for (int kb = 0; kb < kend; kb += BC) {
        {   // stage: K via async global->LDS b128, V transposed via registers
            const int key = tid >> 1;
            const int db  = (tid & 1) * 64;
            const size_t gidx = ((size_t)(b * CS + kb + key)) * CKVD + kvh * CHD + db;
            uint32_t goff = (uint32_t)(gidx * 2u);                        // byte offset
            uint32_t laddr = (uint32_t)(uintptr_t)&sK[key * SKS + db];    // LDS bytes
#pragma unroll
            for (int j = 0; j < 8; ++j) {
                asm volatile("global_load_async_to_lds_b128 %0, %1, %2"
                             :: "v"(laddr + 16u * j), "v"(goff + 16u * j), "s"(kbase64)
                             : "memory");
            }
            const uint16_t* vsrc = V + gidx;
#pragma unroll
            for (int j = 0; j < 32; ++j) {
                uint32_t wv2 = *reinterpret_cast<const uint32_t*>(vsrc + 2 * j);
                int d = db + 2 * j;
                sVT[d * SVS + key]       = (uint16_t)(wv2 & 0xFFFFu);
                sVT[(d + 1) * SVS + key] = (uint16_t)(wv2 >> 16);
            }
        }
        asm volatile("s_wait_asynccnt 0" ::: "memory");
        __syncthreads();

        // scores = Q (16x128) * K^T (128x64): 4 N-tiles x 4 k-steps of WMMA
        f32x8 sc[4];
#pragma unroll
        for (int nt = 0; nt < 4; ++nt)
#pragma unroll
            for (int r = 0; r < 8; ++r) sc[nt][r] = 0.f;

#pragma unroll
        for (int nt = 0; nt < 4; ++nt) {
#pragma unroll
            for (int kd = 0; kd < 4; ++kd) {
                Frag bk;  // B[k=d][n=key] = sK[key][d]: lane row = key, d contiguous
                const uint16_t* base = sK + (nt * 16 + ll) * SKS + kd * 32 + lg * 16;
#pragma unroll
                for (int j = 0; j < 8; ++j)
                    bk.u[j] = *reinterpret_cast<const uint32_t*>(base + 2 * j);
                sc[nt] = __builtin_amdgcn_wmma_f32_16x16x32_bf16(
                    false, qf[kd].bf, false, bk.bf, (short)0, sc[nt], false, false);
            }
        }

        // scale + causal mask + online softmax (row r lives in element r per 16-lane group)
#pragma unroll
        for (int r = 0; r < 8; ++r) {
            const int qi = q0 + lg * 8 + r;
            float mx = -1e30f;
#pragma unroll
            for (int nt = 0; nt < 4; ++nt) {
                float sv = sc[nt][r] * scale;
                int kj = kb + nt * 16 + ll;
                sv = (kj <= qi) ? sv : -1e30f;
                sc[nt][r] = sv;
                mx = fmaxf(mx, sv);
            }
#pragma unroll
            for (int off = 8; off >= 1; off >>= 1) mx = fmaxf(mx, __shfl_xor(mx, off, 32));
            float mnew  = fmaxf(mrow[r], mx);
            float alpha = __expf(mrow[r] - mnew);
            float rs = 0.f;
#pragma unroll
            for (int nt = 0; nt < 4; ++nt) {
                float p = __expf(sc[nt][r] - mnew);
                sc[nt][r] = p;
                rs += p;
            }
#pragma unroll
            for (int off = 8; off >= 1; off >>= 1) rs += __shfl_xor(rs, off, 32);
            lrow[r] = lrow[r] * alpha + rs;
            mrow[r] = mnew;
#pragma unroll
            for (int dt = 0; dt < 8; ++dt) o[dt][r] *= alpha;
        }

        // P: C-layout -> bf16 A-layout via per-wave LDS bounce (intra-wave dscnt wait)
        uint16_t* sp = sP + wid * 16 * SPS;
#pragma unroll
        for (int nt = 0; nt < 4; ++nt)
#pragma unroll
            for (int r = 0; r < 8; ++r)
                sp[(lg * 8 + r) * SPS + nt * 16 + ll] = f2bf(sc[nt][r]);
        asm volatile("s_wait_dscnt 0" ::: "memory");

        Frag pf[2];
#pragma unroll
        for (int kt = 0; kt < 2; ++kt) {
            const uint16_t* base = sp + ll * SPS + kt * 32 + lg * 16;
#pragma unroll
            for (int j = 0; j < 8; ++j) {
                int off = 2 * j;
                pf[kt].u[j] = *reinterpret_cast<const uint32_t*>(base + off);
            }
        }
        // O += P (16x64) * V (64x128): 8 d-tiles x 2 k-steps
#pragma unroll
        for (int dt = 0; dt < 8; ++dt) {
#pragma unroll
            for (int kt = 0; kt < 2; ++kt) {
                Frag bv;  // B[k=key][n=d] = sVT[d][key]: lane row = d, key contiguous
                const uint16_t* base = sVT + (dt * 16 + ll) * SVS + kt * 32 + lg * 16;
#pragma unroll
                for (int j = 0; j < 8; ++j)
                    bv.u[j] = *reinterpret_cast<const uint32_t*>(base + 2 * j);
                o[dt] = __builtin_amdgcn_wmma_f32_16x16x32_bf16(
                    false, pf[kt].bf, false, bv.bf, (short)0, o[dt], false, false);
            }
        }
        __syncthreads();
    }

    // normalize and store bf16 into [B*S, H*128] for the output projection GEMM
#pragma unroll
    for (int r = 0; r < 8; ++r) {
        float inv = 1.f / lrow[r];
        const size_t obase = ((size_t)(b * CS + q0 + lg * 8 + r)) * CQD + h * CHD;
#pragma unroll
        for (int dt = 0; dt < 8; ++dt)
            O[obase + dt * 16 + ll] = f2bf(o[dt][r] * inv);
    }
}

// ---------------------------------------------------------------- launch
extern "C" void kernel_launch(void* const* d_in, const int* in_sizes, int n_in,
                              void* d_out, int out_size, void* d_ws, size_t ws_size,
                              hipStream_t stream)
{
    (void)in_sizes; (void)n_in; (void)out_size; (void)ws_size;
    const float* x    = (const float*)d_in[0];
    const float* fcos = (const float*)d_in[1];
    const float* fsin = (const float*)d_in[2];
    const float* wq   = (const float*)d_in[3];
    const float* wk   = (const float*)d_in[4];
    const float* wv   = (const float*)d_in[5];
    const float* wo   = (const float*)d_in[6];
    float* out = (float*)d_out;

    // workspace carve-up (bf16 buffers), total ~132 MB
    char* ws = (char*)d_ws;
    size_t off = 0;
    uint16_t* xb  = (uint16_t*)(ws + off); off += (size_t)CM  * CD   * 2;
    uint16_t* wqb = (uint16_t*)(ws + off); off += (size_t)CD  * CQD  * 2;
    uint16_t* wkb = (uint16_t*)(ws + off); off += (size_t)CD  * CKVD * 2;
    uint16_t* wvb = (uint16_t*)(ws + off); off += (size_t)CD  * CKVD * 2;
    uint16_t* wob = (uint16_t*)(ws + off); off += (size_t)CQD * CD   * 2;
    uint16_t* qb  = (uint16_t*)(ws + off); off += (size_t)CM  * CQD  * 2;
    uint16_t* kb  = (uint16_t*)(ws + off); off += (size_t)CM  * CKVD * 2;
    uint16_t* vb  = (uint16_t*)(ws + off); off += (size_t)CM  * CKVD * 2;
    uint16_t* ob  = (uint16_t*)(ws + off);

    cvt_kernel<<<(CM * CD)  / 4 / 256, 256, 0, stream>>>(x,  xb,  CM * CD);
    cvt_kernel<<<(CD * CQD) / 4 / 256, 256, 0, stream>>>(wq, wqb, CD * CQD);
    cvt_kernel<<<(CD * CKVD)/ 4 / 256, 256, 0, stream>>>(wk, wkb, CD * CKVD);
    cvt_kernel<<<(CD * CKVD)/ 4 / 256, 256, 0, stream>>>(wv, wvb, CD * CKVD);
    cvt_kernel<<<(CQD * CD) / 4 / 256, 256, 0, stream>>>(wo, wob, CQD * CD);

    // QKV projections (bf16 out)
    gemm_kernel<false><<<dim3(CQD / 128,  CM / 128), 256, 0, stream>>>(xb, wqb, qb, CM, CQD,  CD);
    gemm_kernel<false><<<dim3(CKVD / 128, CM / 128), 256, 0, stream>>>(xb, wkb, kb, CM, CKVD, CD);
    gemm_kernel<false><<<dim3(CKVD / 128, CM / 128), 256, 0, stream>>>(xb, wvb, vb, CM, CKVD, CD);

    // RoPE on q and k (in-place)
    rope_kernel<<<(CB * CS * CNH  * 64) / 256, 256, 0, stream>>>(qb, fcos, fsin, CNH);
    rope_kernel<<<(CB * CS * CNKV * 64) / 256, 256, 0, stream>>>(kb, fcos, fsin, CNKV);

    // causal flash attention with GQA
    flash_kernel<<<dim3(CS / 64, CNH, CB), dim3(128), 0, stream>>>(qb, kb, vb, ob);

    // output projection (f32 out)
    gemm_kernel<true><<<dim3(CD / 128, CM / 128), 256, 0, stream>>>(ob, wob, out, CM, CD, CQD);
}
